// Mask_based_WSM_74440373174558
// MI455X (gfx1250) — matrix operand order repaired and checked by reading, not compiled, not run on hardware
//
#include <hip/hip_runtime.h>
#include <hip/hip_bf16.h>
#include <math.h>

typedef __attribute__((ext_vector_type(16))) _Float16 v16h;
typedef __attribute__((ext_vector_type(8)))  float    v8f;
typedef __attribute__((ext_vector_type(4)))  float    fvec4;  // native vec for NT builtins

#define NBINS     256
#define PER_BATCH (1024 * 1024)

// ---------------------------------------------------------------------------
// Kernel 0: zero the workspace (hist bins + flags). ws is poisoned by harness.
// ---------------------------------------------------------------------------
__global__ void zero_ws_kernel(unsigned* __restrict__ hist, float* __restrict__ flags,
                               int nbatch) {
    int n = nbatch * NBINS;
    for (int i = threadIdx.x; i < n; i += blockDim.x) hist[i] = 0u;
    for (int i = threadIdx.x; i < nbatch; i += blockDim.x) flags[i] = 0.0f;
}

// ---------------------------------------------------------------------------
// Kernel 1: per-batch 256-bin histogram of mi = irr*255 over range (0, 255).
// jnp.histogram bin width = 255/256, so bin = floor(v * 256/255), clamped.
// b128 non-temporal loads (streaming, no reuse); LDS histogram per block
// (ds_add atomics), then flushed to global per-batch hist with atomics.
// ---------------------------------------------------------------------------
__global__ void hist_kernel(const float* __restrict__ irr,
                            unsigned* __restrict__ hist) {
    __shared__ unsigned sh[NBINS];
    const int b = blockIdx.y;
    for (int j = threadIdx.x; j < NBINS; j += blockDim.x) sh[j] = 0u;
    __syncthreads();

    const fvec4* img = (const fvec4*)(irr + (size_t)b * PER_BATCH);
    const int nvec   = PER_BATCH / 4;
    const int stride = gridDim.x * blockDim.x;
    for (int i = blockIdx.x * blockDim.x + threadIdx.x; i < nvec; i += stride) {
        fvec4 x = __builtin_nontemporal_load(&img[i]);    // global_load_b128 NT
#pragma unroll
        for (int c = 0; c < 4; ++c) {
            float v = x[c] * 255.0f;
            if (v >= 0.0f && v <= 255.0f) {
                int bin = (int)(v * (256.0f / 255.0f));
                bin = bin > 255 ? 255 : bin;
                atomicAdd(&sh[bin], 1u);                  // ds_add_u32
            }
        }
    }
    __syncthreads();
    for (int j = threadIdx.x; j < 256; j += blockDim.x) {
        unsigned c = sh[j];
        if (c) atomicAdd(&hist[b * NBINS + j], c);        // global_atomic_add_u32
    }
}

// ---------------------------------------------------------------------------
// Kernel 2: one wave32 per batch. mask_output = dist @ hist via WMMA
// (v_wmma_f32_16x16x32_f16). B columns are all hist (broadcast), so every
// column of D equals the matvec result regardless of N mapping.
// A layout (16-bit A 16x32, ISA 7.12.2): lane (m,g), half h -> K =
//   (h>=8 ? 16:0) + 2*((h>>1)&3) + (h&1) + 8*g  with v=h>>1
// B layout (16-bit B 32x16, matches A): half h -> K = (h>=8 ? 8+h : h) + 8*g
// C/D layout: VGPR r, lane group g -> row M = r + 8*g.
// Then flag[b] = (max over occupied bins of mask_output) > 0.
// ---------------------------------------------------------------------------
__global__ __launch_bounds__(32) void matvec_wmma_kernel(
        const unsigned* __restrict__ hist, float* __restrict__ flags) {
    __shared__ float sh[NBINS];
    const int b    = blockIdx.x;
    const int lane = threadIdx.x;      // 0..31 (wave32)
    const int g    = lane >> 4;        // lane group
    const int m    = lane & 15;

    const unsigned* h = hist + b * NBINS;
    for (int j = lane; j < NBINS; j += 32) {
        // clamp so f16 conversion stays finite (avoids 0*inf NaN in WMMA);
        // preserves zero-vs-nonzero which is all that feeds the output
        sh[j] = fminf((float)h[j], 60000.0f);
    }
    __syncthreads();

    float localmax = 0.0f;
    for (int mb = 0; mb < 16; ++mb) {              // 16 row-blocks of dist
        v8f acc = {};
        const int row = mb * 16 + m;               // this lane's A row
#pragma unroll
        for (int kc = 0; kc < 8; ++kc) {           // K = 256 in chunks of 32
            const int kbase = kc * 32;
            v16h a, bb;
#pragma unroll
            for (int hh = 0; hh < 16; ++hh) {
                const int v   = hh >> 1;
                const int bit = hh & 1;
                const int ka  = ((v < 4) ? 0 : 16) + 2 * (v & 3) + bit + 8 * g + kbase;
                a[hh] = (_Float16)fabsf((float)(row - ka));   // dist[row][ka]
                const int kb = ((hh < 8) ? hh : 8 + hh) + 8 * g + kbase;
                bb[hh] = (_Float16)sh[kb];                    // hist broadcast
            }
            acc = __builtin_amdgcn_wmma_f32_16x16x32_f16(
                false, a, false, bb, (short)0, acc, false, false);
        }
        // all columns equal; lane holds rows M = r + 8g of this block
#pragma unroll
        for (int r = 0; r < 8; ++r) {
            const int j = mb * 16 + r + 8 * g;
            float mv = (sh[j] > 0.0f) ? acc[r] : 0.0f;  // only occupied bins
            localmax = fmaxf(localmax, mv);
        }
    }
    // wave32 max-reduction
    for (int off = 16; off >= 1; off >>= 1)
        localmax = fmaxf(localmax, __shfl_xor(localmax, off, 32));
    if (lane == 0) flags[b] = (localmax > 0.0f) ? 1.0f : 0.0f;
}

// ---------------------------------------------------------------------------
// Kernel 3: bandwidth-bound fused elementwise pass. All traffic is streaming
// (read 64MB, write 128MB, zero reuse ~= full 192MB L2), so use non-temporal
// b128 loads/stores to keep L2 out of the way of HBM streaming.
// m = flag ? (x*255)/255 : 0 ; softmax over {m, 1-m}:
//   out0 = e^m / (e^m + e^(1-m)), out1 = e^(1-m) / (e^m + e^(1-m))
// ---------------------------------------------------------------------------
__global__ void fuse_kernel(const float* __restrict__ irr,
                            const float* __restrict__ flags,
                            float* __restrict__ out, int total) {
    const int i4 = blockIdx.x * blockDim.x + threadIdx.x;
    const int i  = i4 * 4;
    if (i >= total) return;
    const int bt = i / PER_BATCH;                  // PER_BATCH % 4 == 0
    const float f = flags[bt];

    fvec4 x = __builtin_nontemporal_load((const fvec4*)(irr + i));
    fvec4 o0, o1;
#pragma unroll
    for (int c = 0; c < 4; ++c) {
        float xv = x[c];
        float mv = f * ((xv * 255.0f) / 255.0f);   // masked/255 (or 0)
        float e0 = expf(mv);
        float e1 = expf(1.0f - mv);
        float inv = 1.0f / (e0 + e1);
        o0[c] = e0 * inv;
        o1[c] = e1 * inv;
    }
    __builtin_nontemporal_store(o0, (fvec4*)(out + i));          // msks[0]
    __builtin_nontemporal_store(o1, (fvec4*)(out + total + i));  // msks[1]
}

// ---------------------------------------------------------------------------
extern "C" void kernel_launch(void* const* d_in, const int* in_sizes, int n_in,
                              void* d_out, int out_size, void* d_ws, size_t ws_size,
                              hipStream_t stream) {
    const float* irr = (const float*)d_in[0];      // image_irr (16,1,1024,1024)
    // d_in[1] (image_vis) is never used by the reference computation
    float* out = (float*)d_out;

    const int total  = in_sizes[0];                // 16 * 1024 * 1024
    const int nbatch = total / PER_BATCH;          // 16

    unsigned* hist  = (unsigned*)d_ws;                               // nbatch*256 u32
    float*    flags = (float*)((char*)d_ws + (size_t)nbatch * NBINS * sizeof(unsigned));

    zero_ws_kernel<<<1, 256, 0, stream>>>(hist, flags, nbatch);
    hist_kernel<<<dim3(64, nbatch), 256, 0, stream>>>(irr, hist);
    matvec_wmma_kernel<<<nbatch, 32, 0, stream>>>(hist, flags);

    const int n4 = total / 4;
    fuse_kernel<<<(n4 + 255) / 256, 256, 0, stream>>>(irr, flags, out, total);
}
